// Informer_46239617909409
// MI455X (gfx1250) — matrix operand order repaired and checked by reading, not compiled, not run on hardware
//
#include <hip/hip_runtime.h>

#define D_MODEL 128
#define D_FF    512
#define NW      4           // waves per block
#define BM      (NW * 16)   // 64 rows per block
#define HS      132         // f32 LDS row stride (pad)
#define VS      136         // bf16 LDS row stride (pad)
#define GS      40          // gelu chunk bf16 row stride (pad)

typedef __attribute__((ext_vector_type(16))) __bf16       v16bf;
typedef __attribute__((ext_vector_type(8)))  float        v8f;
typedef __attribute__((ext_vector_type(4)))  float        v4f;
typedef __attribute__((ext_vector_type(4)))  unsigned int v4u;

union FragB { v16bf v; v4u q[2]; };

__device__ __forceinline__ v8f wmma_bf16(v16bf a, v16bf b, v8f c) {
    return __builtin_amdgcn_wmma_f32_16x16x32_bf16(false, a, false, b, (short)0, c,
                                                   false, false);
}

// Fast tanh-GELU: x*sigmoid(2u) = x*(1 - 1/(1+e^{2u})), u = 0.79788456*(x+0.044715x^3)
// Tails are exact: e^{2u}->inf => g->x ; e^{2u}->0 => g->0. Max abs err ~3e-4.
__device__ __forceinline__ float gelu_f(float x) {
    float x2 = x * x;
    float u  = x * (0.79788456080286536f + 0.03567740813636141f * x2);
    float t  = __builtin_amdgcn_exp2f(u * 2.8853900817779268f);   // e^{2u}
    return x - x * __builtin_amdgcn_rcpf(1.0f + t);
}

// A fragment (16x32 bf16) built from f32 LDS slab (rows m0..m0+15 of this wave)
__device__ __forceinline__ v16bf load_a_f32(const float* slab, int kc, int lane) {
    int half = lane >> 4, mr = lane & 15;
    const float* p = slab + mr * HS + kc + half * 8;
    v4f a0 = *(const v4f*)(p);
    v4f a1 = *(const v4f*)(p + 4);
    v4f a2 = *(const v4f*)(p + 16);
    v4f a3 = *(const v4f*)(p + 20);
    v16bf f;
    f[0]  = (__bf16)a0.x; f[1]  = (__bf16)a0.y; f[2]  = (__bf16)a0.z; f[3]  = (__bf16)a0.w;
    f[4]  = (__bf16)a1.x; f[5]  = (__bf16)a1.y; f[6]  = (__bf16)a1.z; f[7]  = (__bf16)a1.w;
    f[8]  = (__bf16)a2.x; f[9]  = (__bf16)a2.y; f[10] = (__bf16)a2.z; f[11] = (__bf16)a2.w;
    f[12] = (__bf16)a3.x; f[13] = (__bf16)a3.y; f[14] = (__bf16)a3.z; f[15] = (__bf16)a3.w;
    return f;
}

// A fragment from bf16 LDS slab
__device__ __forceinline__ v16bf load_a_bf16(const __bf16* slab, int stride,
                                             int kc, int lane) {
    int half = lane >> 4, mr = lane & 15;
    const __bf16* p = slab + mr * stride + kc + half * 8;
    FragB f;
    f.q[0] = *(const v4u*)(p);
    f.q[1] = *(const v4u*)(p + 16);
    return f.v;
}

// B fragment (32x16) from row-major bf16 weight W[N][K] in global (hits L2)
__device__ __forceinline__ v16bf load_b_glb(const __bf16* W, int K,
                                            int n0, int kc, int lane) {
    int half = lane >> 4;
    const __bf16* p = W + (size_t)(n0 + (lane & 15)) * K + kc + half * 8;
    FragB f;
    f.q[0] = *(const v4u*)(p);
    f.q[1] = *(const v4u*)(p + 16);
    return f.v;
}

// LayerNorm over 16 rows of the wave's f32 slab (128 cols), wave32 reduction
__device__ __forceinline__ void ln_rows(float* slab, int lane,
                                        const float* g, const float* b) {
    float g0 = g[lane], g1 = g[lane + 32], g2 = g[lane + 64], g3 = g[lane + 96];
    float c0 = b[lane], c1 = b[lane + 32], c2 = b[lane + 64], c3 = b[lane + 96];
    for (int rr = 0; rr < 16; ++rr) {
        float* row = slab + rr * HS;
        float v0 = row[lane], v1 = row[lane + 32], v2 = row[lane + 64], v3 = row[lane + 96];
        float s  = v0 + v1 + v2 + v3;
        float s2 = v0 * v0 + v1 * v1 + v2 * v2 + v3 * v3;
        #pragma unroll
        for (int off = 16; off; off >>= 1) {
            s  += __shfl_xor(s,  off, 32);
            s2 += __shfl_xor(s2, off, 32);
        }
        float mu   = s * 0.0078125f;
        float rstd = rsqrtf(s2 * 0.0078125f - mu * mu + 1e-5f);
        row[lane]      = (v0 - mu) * rstd * g0 + c0;
        row[lane + 32] = (v1 - mu) * rstd * g1 + c1;
        row[lane + 64] = (v2 - mu) * rstd * g2 + c2;
        row[lane + 96] = (v3 - mu) * rstd * g3 + c3;
    }
}

__global__ void cvt_f32_bf16(const float* __restrict__ src,
                             __bf16* __restrict__ dst, int n) {
    int i = blockIdx.x * blockDim.x + threadIdx.x;
    if (i < n) dst[i] = (__bf16)src[i];
}

__global__ __launch_bounds__(128) void informer_fwd(
    const float* __restrict__ x,       const float* __restrict__ tfeat,
    const float* __restrict__ conv_w,  const float* __restrict__ time_w,
    const float* __restrict__ pbv,     const float* __restrict__ pbo,
    const float* __restrict__ pb1,     const float* __restrict__ pb2,
    const float* __restrict__ ln1_g,   const float* __restrict__ ln1_b,
    const float* __restrict__ ln2_g,   const float* __restrict__ ln2_b,
    const float* __restrict__ lnf_g,   const float* __restrict__ lnf_b,
    const float* __restrict__ fc_w,    const float* __restrict__ fc_b,
    const __bf16* __restrict__ wv16,   const __bf16* __restrict__ wo16,
    const __bf16* __restrict__ w116,   const __bf16* __restrict__ w216,
    float* __restrict__ out)
{
    __shared__ float  sh_h[BM * HS];   // f32 hidden state
    __shared__ __bf16 sh_v[BM * VS];   // bf16 GEMM intermediate
    __shared__ __bf16 sh_g[BM * GS];   // bf16 gelu chunk (16x32 per wave)

    const int lane = threadIdx.x & 31;
    const int wid  = threadIdx.x >> 5;
    const int m0   = wid * 16;                 // wave's row slab in LDS
    const int rowbase = blockIdx.x * BM + m0;  // global row base of slab
    const int half = lane >> 4;
    const int nl   = lane & 15;

    float*  slab  = sh_h + m0 * HS;
    __bf16* vslab = sh_v + m0 * VS;
    __bf16* gslab = sh_g + m0 * GS;

    // ---- Embedding: h = x @ sum_k(conv_w).T + pe + time @ time_w.T ----
    {
        float wsum[4][17], twl[4][4];
        #pragma unroll
        for (int j = 0; j < 4; ++j) {
            int d = lane + 32 * j;
            #pragma unroll
            for (int c = 0; c < 17; ++c) {
                const float* cw = conv_w + (d * 17 + c) * 3;
                wsum[j][c] = cw[0] + cw[1] + cw[2];
            }
            #pragma unroll
            for (int t = 0; t < 4; ++t) twl[j][t] = time_w[d * 4 + t];
        }
        for (int rr = 0; rr < 16; ++rr) {
            int grow = rowbase + rr;
            float xr[17], tr[4];
            #pragma unroll
            for (int c = 0; c < 17; ++c) xr[c] = x[grow * 17 + c];
            #pragma unroll
            for (int t = 0; t < 4; ++t) tr[t] = tfeat[grow * 4 + t];
            #pragma unroll
            for (int j = 0; j < 4; ++j) {
                int d = lane + 32 * j;
                float acc = (d & 1) ? 1.0f : 0.0f;  // positional encoding
                #pragma unroll
                for (int c = 0; c < 17; ++c) acc += xr[c] * wsum[j][c];
                #pragma unroll
                for (int t = 0; t < 4; ++t) acc += tr[t] * twl[j][t];
                slab[rr * HS + d] = acc;
            }
        }
    }

    // ---- 3 encoder layers (seq len 1 => attention == Wo(Wv(h)) ) ----
    for (int l = 0; l < 3; ++l) {
        const __bf16* Wvl = wv16 + l * D_MODEL * D_MODEL;
        const __bf16* Wol = wo16 + l * D_MODEL * D_MODEL;
        const __bf16* W1l = w116 + l * D_FF * D_MODEL;
        const __bf16* W2l = w216 + l * D_MODEL * D_FF;

        // Stage 1: v = h @ Wv.T + bv  -> bf16 in sh_v  (B tiles double-buffered)
        {
            v16bf ah[4];
            #pragma unroll
            for (int kk = 0; kk < 4; ++kk) ah[kk] = load_a_f32(slab, kk * 32, lane);
            v16bf bb[2][4];
            #pragma unroll
            for (int kk = 0; kk < 4; ++kk)
                bb[0][kk] = load_b_glb(Wvl, D_MODEL, 0, kk * 32, lane);
            #pragma unroll
            for (int nt = 0; nt < 8; ++nt) {
                const int cur = nt & 1;
                if (nt < 7) {
                    #pragma unroll
                    for (int kk = 0; kk < 4; ++kk)
                        bb[cur ^ 1][kk] = load_b_glb(Wvl, D_MODEL, (nt + 1) * 16, kk * 32, lane);
                }
                int nc = nt * 16 + nl;
                float bias = pbv[l * D_MODEL + nc];
                v8f c;
                #pragma unroll
                for (int r = 0; r < 8; ++r) c[r] = bias;
                #pragma unroll
                for (int kk = 0; kk < 4; ++kk) c = wmma_bf16(ah[kk], bb[cur][kk], c);
                __bf16* vp = vslab + half * 8 * VS + nc;
                #pragma unroll
                for (int r = 0; r < 8; ++r) vp[r * VS] = (__bf16)c[r];
            }
        }

        // Stage 2: h = h + v @ Wo.T + bo ; then LN1  (B tiles double-buffered)
        {
            v16bf av[4];
            #pragma unroll
            for (int kk = 0; kk < 4; ++kk) av[kk] = load_a_bf16(vslab, VS, kk * 32, lane);
            v16bf bb[2][4];
            #pragma unroll
            for (int kk = 0; kk < 4; ++kk)
                bb[0][kk] = load_b_glb(Wol, D_MODEL, 0, kk * 32, lane);
            #pragma unroll
            for (int nt = 0; nt < 8; ++nt) {
                const int cur = nt & 1;
                if (nt < 7) {
                    #pragma unroll
                    for (int kk = 0; kk < 4; ++kk)
                        bb[cur ^ 1][kk] = load_b_glb(Wol, D_MODEL, (nt + 1) * 16, kk * 32, lane);
                }
                int nc = nt * 16 + nl;
                float bias = pbo[l * D_MODEL + nc];
                float* hp = slab + half * 8 * HS + nc;
                v8f c;
                #pragma unroll
                for (int r = 0; r < 8; ++r) c[r] = hp[r * HS] + bias;
                #pragma unroll
                for (int kk = 0; kk < 4; ++kk) c = wmma_bf16(av[kk], bb[cur][kk], c);
                #pragma unroll
                for (int r = 0; r < 8; ++r) hp[r * HS] = c[r];
            }
            ln_rows(slab, lane, ln1_g + l * D_MODEL, ln1_b + l * D_MODEL);
        }

        // Stage 3: FFN fused chunk-wise over D_FF:
        //   acc = h + b2 ; per 32-wide chunk: g = GELU(h@W1.T+b1)[:,chunk];
        //   acc += g @ W2[:,chunk].T   (W2 B tiles hoisted above GELU to hide latency)
        {
            v16bf af[4];
            #pragma unroll
            for (int kk = 0; kk < 4; ++kk) af[kk] = load_a_f32(slab, kk * 32, lane);

            v8f acc[8];
            #pragma unroll
            for (int nt = 0; nt < 8; ++nt) {
                int nc = nt * 16 + nl;
                float bias = pb2[l * D_MODEL + nc];
                const float* hp = slab + half * 8 * HS + nc;
                #pragma unroll
                for (int r = 0; r < 8; ++r) acc[nt][r] = hp[r * HS] + bias;
            }

            for (int fc0 = 0; fc0 < D_FF; fc0 += 32) {
                // batch-load W1 B tiles for both 16-wide columns of this chunk
                v16bf bw1[2][4];
                #pragma unroll
                for (int t = 0; t < 2; ++t)
                    #pragma unroll
                    for (int kk = 0; kk < 4; ++kk)
                        bw1[t][kk] = load_b_glb(W1l, D_MODEL, fc0 + t * 16, kk * 32, lane);

                v8f cc[2];
                #pragma unroll
                for (int t = 0; t < 2; ++t) {
                    float bias = pb1[l * D_FF + fc0 + t * 16 + nl];
                    #pragma unroll
                    for (int r = 0; r < 8; ++r) cc[t][r] = bias;
                    #pragma unroll
                    for (int kk = 0; kk < 4; ++kk)
                        cc[t] = wmma_bf16(af[kk], bw1[t][kk], cc[t]);
                }

                // prefetch all W2 B tiles for this chunk; latency hides under GELU
                v16bf bw2[8];
                #pragma unroll
                for (int nt = 0; nt < 8; ++nt)
                    bw2[nt] = load_b_glb(W2l, D_FF, nt * 16, fc0, lane);

                #pragma unroll
                for (int t = 0; t < 2; ++t) {
                    __bf16* gp = gslab + half * 8 * GS + t * 16 + nl;
                    #pragma unroll
                    for (int r = 0; r < 8; ++r) gp[r * GS] = (__bf16)gelu_f(cc[t][r]);
                }
                v16bf a2 = load_a_bf16(gslab, GS, 0, lane);
                #pragma unroll
                for (int nt = 0; nt < 8; ++nt)
                    acc[nt] = wmma_bf16(a2, bw2[nt], acc[nt]);
            }

            #pragma unroll
            for (int nt = 0; nt < 8; ++nt) {
                float* hp = slab + half * 8 * HS + nt * 16 + nl;
                #pragma unroll
                for (int r = 0; r < 8; ++r) hp[r * HS] = acc[nt][r];
            }
            ln_rows(slab, lane, ln2_g + l * D_MODEL, ln2_b + l * D_MODEL);
        }
    }

    // ---- Final LN + head: out = h @ fc_w.T + fc_b ----
    ln_rows(slab, lane, lnf_g, lnf_b);
    float f0 = fc_w[lane], f1 = fc_w[lane + 32], f2 = fc_w[lane + 64], f3 = fc_w[lane + 96];
    float fb = fc_b[0];
    for (int rr = 0; rr < 16; ++rr) {
        const float* row = slab + rr * HS;
        float s = row[lane] * f0 + row[lane + 32] * f1 +
                  row[lane + 64] * f2 + row[lane + 96] * f3;
        #pragma unroll
        for (int off = 16; off; off >>= 1) s += __shfl_xor(s, off, 32);
        if (lane == 0) out[rowbase + rr] = s + fb;
    }
}

extern "C" void kernel_launch(void* const* d_in, const int* in_sizes, int n_in,
                              void* d_out, int out_size, void* d_ws, size_t ws_size,
                              hipStream_t stream) {
    (void)in_sizes; (void)n_in; (void)out_size; (void)ws_size;
    const float* x      = (const float*)d_in[0];
    const float* tfeat  = (const float*)d_in[1];
    const float* conv_w = (const float*)d_in[2];
    const float* time_w = (const float*)d_in[3];
    // d_in[4..7] = Wq,bq,Wk,bk : unused by the reference computation
    const float* Wv     = (const float*)d_in[8];
    const float* bv     = (const float*)d_in[9];
    const float* Wo     = (const float*)d_in[10];
    const float* bo     = (const float*)d_in[11];
    const float* W1     = (const float*)d_in[12];
    const float* b1     = (const float*)d_in[13];
    const float* W2     = (const float*)d_in[14];
    const float* b2     = (const float*)d_in[15];
    const float* ln1_g  = (const float*)d_in[16];
    const float* ln1_b  = (const float*)d_in[17];
    const float* ln2_g  = (const float*)d_in[18];
    const float* ln2_b  = (const float*)d_in[19];
    const float* lnf_g  = (const float*)d_in[20];
    const float* lnf_b  = (const float*)d_in[21];
    const float* fc_w   = (const float*)d_in[22];
    const float* fc_b   = (const float*)d_in[23];
    float* out = (float*)d_out;

    // bf16 weight staging in workspace (converted every call; deterministic)
    __bf16* ws   = (__bf16*)d_ws;
    __bf16* wv16 = ws;                 // 3*128*128 = 49152
    __bf16* wo16 = ws + 49152;         // 49152
    __bf16* w116 = ws + 98304;         // 3*512*128 = 196608
    __bf16* w216 = ws + 294912;        // 196608

    cvt_f32_bf16<<<(49152  + 255) / 256, 256, 0, stream>>>(Wv, wv16, 49152);
    cvt_f32_bf16<<<(49152  + 255) / 256, 256, 0, stream>>>(Wo, wo16, 49152);
    cvt_f32_bf16<<<(196608 + 255) / 256, 256, 0, stream>>>(W1, w116, 196608);
    cvt_f32_bf16<<<(196608 + 255) / 256, 256, 0, stream>>>(W2, w216, 196608);

    informer_fwd<<<65536 / BM, NW * 32, 0, stream>>>(
        x, tfeat, conv_w, time_w, bv, bo, b1, b2,
        ln1_g, ln1_b, ln2_g, ln2_b, lnf_g, lnf_b, fc_w, fc_b,
        wv16, wo16, w116, w216, out);
}
